// Parametric_LSTM_47064251630155
// MI455X (gfx1250) — matrix operand order repaired and checked by reading, not compiled, not run on hardware
//
#include <hip/hip_runtime.h>
#include <hip/hip_bf16.h>
#include <stdint.h>

// ---------------------------------------------------------------------------
// Parametric LSTM for MI455X (gfx1250, wave32, WMMA).
//
//  * Recurrence independent per batch row -> 8 WGs of 16 batch rows,
//    16 waves per WG, wave w owns unit tile [16w, 16w+16).
//  * Fused per-step GEMM: z = [h | X_t | S_t] @ [W; U; S_param] + b,
//    K = 416, N = 1024, via v_wmma_f32_16x16x32_bf16 (fp32 accum).
//  * Weights pre-swizzled into the exact 32x16 bf16 B-fragment layout ->
//    two coalesced global_load_b128 per fragment, L2-resident (852 KB).
//  * A (16x416 bf16) staged in LDS -> two ds_load_b128 per A fragment.
//  * An opaque (asm volatile) zero offset is added to the weight pointer
//    inside the t loop so the compiler cannot hoist the 104 B-fragment
//    loads out of the 512-step loop (which previously caused a full
//    VGPR spill and scratch_load_b128 reloads every step).
// ---------------------------------------------------------------------------

typedef __attribute__((ext_vector_type(16))) __bf16 v16bf;
typedef __attribute__((ext_vector_type(8)))  float  v8f;

#define T_STEPS     512
#define D_IN        128
#define D_P         32
#define UNITS       256
#define NCOL        1024        // 4*units
#define K_TOT       416         // units + D_in + D_p
#define K_TILES     13          // 416 / 32
#define N_TILES     64          // 1024 / 16
#define A_STRIDE    424         // padded A row stride (bf16), 848B (16B aligned)
#define T_OUT_START 384         // T - num_outputs

__device__ __forceinline__ unsigned short f2bf(float f) {
  union { float f; unsigned int u; } x; x.f = f;
  unsigned int r = x.u + 0x7FFFu + ((x.u >> 16) & 1u);   // round-nearest-even
  return (unsigned short)(r >> 16);
}

__device__ __forceinline__ float sigm(float x) {
  return 1.0f / (1.0f + __expf(-x));
}

__device__ __forceinline__ float ftanh(float x) {
  float a = fabsf(x);
  float e = __expf(-2.0f * a);            // e in (0,1], no overflow
  float t = (1.0f - e) / (1.0f + e);
  return copysignf(t, x);
}

// ---------------------------------------------------------------------------
// Weight swizzle: pack [W(256x1024); U(128x1024); S_param(32x1024)] (fp32,
// row-major) into bf16 blocks matching the 32x16 B-fragment register layout:
//   block bi = kt*64 + nt  (1 KB each)
//   lane l, chunk c (c=0: VGPR0-3, c=1: VGPR4-7):
//     16 bytes = B[ kt*32 + 16*(l/16) + 8*c + j ][ nt*16 + l%16 ], j=0..7
// ---------------------------------------------------------------------------
__global__ void Parametric_LSTM_weight_swizzle(const float* __restrict__ W,
                                               const float* __restrict__ U,
                                               const float* __restrict__ Sp,
                                               unsigned short* __restrict__ Wq) {
  const int idx = blockIdx.x * 256 + threadIdx.x;    // < 416*1024 = 425984
  const int bi  = idx >> 9;                          // block (512 bf16 each)
  const int o   = idx & 511;
  const int kt  = bi >> 6;
  const int nt  = bi & 63;
  const int c   = o >> 8;
  const int rem = o & 255;
  const int l   = rem >> 3;
  const int j   = rem & 7;
  const int k   = kt * 32 + ((l >> 4) << 4) + (c << 3) + j;   // 0..415
  const int n   = (nt << 4) + (l & 15);                        // 0..1023
  float v;
  if (k < UNITS)             v = W [(size_t)k * NCOL + n];
  else if (k < UNITS + D_IN) v = U [(size_t)(k - UNITS) * NCOL + n];
  else                       v = Sp[(size_t)(k - UNITS - D_IN) * NCOL + n];
  Wq[idx] = f2bf(v);
}

// ---------------------------------------------------------------------------
union Frag { v16bf v; uint4 q[2]; };

__global__ __launch_bounds__(512, 1)
void Parametric_LSTM_scan(const float* __restrict__ X,
                          const float* __restrict__ S,
                          const float* __restrict__ bvec,
                          const unsigned short* __restrict__ Wq,
                          float* __restrict__ out) {
  __shared__ __align__(16) unsigned short Alds[16 * A_STRIDE];

  const int tid  = threadIdx.x;
  const int lane = tid & 31;
  const int wv   = tid >> 5;        // 0..15 : unit tile owned by this wave
  const int half = lane >> 4;       // 0/1
  const int n16  = lane & 15;
  const int b0   = blockIdx.x << 4; // batch row base

  // h region of A (K = 0..255) starts at zero
  for (int i = tid; i < 16 * UNITS; i += 512)
    Alds[(i >> 8) * A_STRIDE + (i & 255)] = 0;

  // per-lane gate biases (column = g*256 + 16*wv + n16), loaded once
  float bb[4];
#pragma unroll
  for (int g = 0; g < 4; ++g) bb[g] = bvec[g * UNITS + wv * 16 + n16];

  v8f c = {};                       // cell state tile (fp32)

  for (int t = 0; t < T_STEPS; ++t) {
    // Opaque zero: forces the weight base to be re-derived every step so
    // B-fragment loads stay inside the t loop (no hoist -> no VGPR spill).
    unsigned int toff;
    asm volatile("s_mov_b32 %0, 0" : "=s"(toff));
    const unsigned short* Wqt = Wq + toff;

    // ---- stage X_t : 16 x 128 fp32 -> bf16 into A[., 256..383] ----
    {
      const int i4 = tid << 2;                     // 4 elems per thread
      const int m = i4 >> 7, d = i4 & 127;
      const float4 xv = *(const float4*)(X + (((size_t)(b0 + m)) * T_STEPS + t) * D_IN + d);
      uint2 pk;
      pk.x = (unsigned int)f2bf(xv.x) | ((unsigned int)f2bf(xv.y) << 16);
      pk.y = (unsigned int)f2bf(xv.z) | ((unsigned int)f2bf(xv.w) << 16);
      *(uint2*)(&Alds[m * A_STRIDE + UNITS + d]) = pk;
    }
    // ---- stage S_t : 16 x 32 fp32 -> bf16 into A[., 384..415] ----
    {
      const int m = tid >> 5, d = tid & 31;
      const float sv = S[(((size_t)(b0 + m)) * T_STEPS + t) * D_P + d];
      Alds[m * A_STRIDE + UNITS + D_IN + d] = f2bf(sv);
    }
    __syncthreads();   // A fully staged (h from prev step + X_t + S_t)

    // ---- z = A @ Wcat for this wave's 4 gate tiles ----
    v8f z[4];
#pragma unroll
    for (int g = 0; g < 4; ++g) z[g] = v8f{};

#pragma unroll
    for (int kt = 0; kt < K_TILES; ++kt) {
      // A fragment (16x32 bf16): lane -> row n16, K chunks at half*16, half*16+8
      Frag a;
      const unsigned short* arow = &Alds[n16 * A_STRIDE + kt * 32 + half * 16];
      a.q[0] = *(const uint4*)(arow);
      a.q[1] = *(const uint4*)(arow + 8);
#pragma unroll
      for (int g = 0; g < 4; ++g) {
        // B fragment: pre-swizzled block -> two coalesced b128 loads
        Frag bf;
        const uint4* bp = (const uint4*)(Wqt + ((kt * N_TILES + (g * 16 + wv)) << 9));
        bf.q[0] = bp[lane];
        bf.q[1] = bp[32 + lane];
        z[g] = __builtin_amdgcn_wmma_f32_16x16x32_bf16(
            false, a.v, false, bf.v, (short)0, z[g], false, false);
      }
    }
    __syncthreads();   // all A reads done; safe to overwrite h region

    // ---- gates, state update, h write-back (C/D layout: VGPR r, lane ->
    //      M = r + 8*half, N = n16) ----
    const bool emit = (t >= T_OUT_START);
#pragma unroll
    for (int r = 0; r < 8; ++r) {
      const float ig = sigm (z[0][r] + bb[0]);
      const float fg = sigm (z[1][r] + bb[1]);
      const float gg = ftanh(z[2][r] + bb[2]);
      const float og = sigm (z[3][r] + bb[3]);
      const float cn = fg * c[r] + ig * gg;
      c[r] = cn;
      const float h = og * ftanh(cn);
      const int M = r + half * 8;
      Alds[M * A_STRIDE + wv * 16 + n16] = f2bf(h);        // h -> A for t+1
      if (emit)
        out[(((size_t)(b0 + M)) * 128 + (t - T_OUT_START)) * UNITS
            + wv * 16 + n16] = h;
    }
    // h writes land before the staging barrier of the next iteration.
  }
}

// ---------------------------------------------------------------------------
extern "C" void kernel_launch(void* const* d_in, const int* in_sizes, int n_in,
                              void* d_out, int out_size, void* d_ws, size_t ws_size,
                              hipStream_t stream) {
  (void)in_sizes; (void)n_in; (void)out_size; (void)ws_size;
  const float* X  = (const float*)d_in[0];   // [128, 512, 128]
  const float* S  = (const float*)d_in[1];   // [128, 512, 32]
  const float* U  = (const float*)d_in[2];   // [128, 1024]
  const float* W  = (const float*)d_in[3];   // [256, 1024]
  const float* Sp = (const float*)d_in[4];   // [32, 1024]
  const float* b  = (const float*)d_in[5];   // [1024]
  // d_in[6] = num_outputs (device scalar) -- shapes are static, value = 128.

  unsigned short* Wq = (unsigned short*)d_ws;  // needs 416*1024*2 = 832 KB

  // 416*1024 = 425984 = 1664 * 256 bf16 elements
  Parametric_LSTM_weight_swizzle<<<1664, 256, 0, stream>>>(W, U, Sp, Wq);
  Parametric_LSTM_scan<<<8, 512, 0, stream>>>(X, S, b, Wq, (float*)d_out);
}
